// SparseConvNeXtBlock_55774445306380
// MI455X (gfx1250) — matrix operand order repaired and checked by reading, not compiled, not run on hardware
//
#include <hip/hip_runtime.h>
#include <hip/hip_bf16.h>
#include <math.h>

typedef __bf16 bf16_t;
typedef __bf16 v16bf __attribute__((ext_vector_type(16)));
typedef __bf16 v8bf  __attribute__((ext_vector_type(8)));
typedef float  v8f   __attribute__((ext_vector_type(8)));
typedef int    v4i32 __attribute__((vector_size(16)));   // matches builtin param type

#define M_TOT 25088   // 32*28*28
#define CDIM  384
#define HIDD  1536
#define LDT   40      // LDS tile row stride in bf16 units (80B, conflict-free)

// ---------------------------------------------------------------------------
// CDNA5 async global->LDS copy (ASYNCcnt path), with synchronous fallback.
// ---------------------------------------------------------------------------
#if defined(__has_builtin) && __has_builtin(__builtin_amdgcn_global_load_async_to_lds_b128)
#define HAVE_ASYNC_LDS 1
__device__ __forceinline__ void copy16(const bf16_t* g, bf16_t* l) {
    __builtin_amdgcn_global_load_async_to_lds_b128(
        (__attribute__((address_space(1))) v4i32*)(void*)g,
        (__attribute__((address_space(3))) v4i32*)(void*)l, 0, 0);
}
#if defined(__has_builtin) && __has_builtin(__builtin_amdgcn_s_wait_asynccnt)
__device__ __forceinline__ void wait_async0() { __builtin_amdgcn_s_wait_asynccnt(0); }
#else
__device__ __forceinline__ void wait_async0() { asm volatile("s_wait_asynccnt 0x0" ::: "memory"); }
#endif
#else
#define HAVE_ASYNC_LDS 0
__device__ __forceinline__ void copy16(const bf16_t* g, bf16_t* l) {
    *(v8bf*)l = *(const v8bf*)g;
}
__device__ __forceinline__ void wait_async0() {}
#endif

// ---------------------------------------------------------------------------
// Depthwise 7x7 reweighted sparse conv.  One block per (b, output row hh).
// Output: conv_out[(b*784 + hh*28 + w)*384 + c], f32 channels-last.
// ---------------------------------------------------------------------------
__global__ __launch_bounds__(256) void k_dwconv(
    const float* __restrict__ x, const int* __restrict__ mask,
    const float* __restrict__ dw_w, const float* __restrict__ dw_b,
    float* __restrict__ cw)
{
    __shared__ float xs[32][7][29];   // [c_local][ky][w], padded (stride 203 words, odd)
    __shared__ float wsh[32][49];     // stride 49 (odd) -> conflict free
    __shared__ float act[7][28];

    const int b  = blockIdx.x;
    const int hh = blockIdx.y;
    const int t  = threadIdx.x;

    for (int i = t; i < 7 * 28; i += 256) {
        int yy = i / 28, w = i % 28;
        int y = hh - 3 + yy;
        float a = 0.f;
        if (y >= 0 && y < 28) a = (float)mask[b * 49 + (y >> 2) * 7 + (w >> 2)];
        act[yy][w] = a;
    }

    const int c_local = t & 31;
    const int wsub    = t >> 5;

    for (int c0 = 0; c0 < CDIM; c0 += 32) {
        for (int i = t; i < 32 * 49; i += 256)
            wsh[i / 49][i % 49] = dw_w[(c0 + i / 49) * 49 + (i % 49)];
        for (int i = t; i < 32 * 7 * 28; i += 256) {
            int w = i % 28, yy = (i / 28) % 7, cl = i / 196;
            int y = hh - 3 + yy;
            float v = 0.f;
            if (y >= 0 && y < 28)
                v = x[(((size_t)b * CDIM + c0 + cl) * 28 + y) * 28 + w];
            xs[cl][yy][w] = v;
        }
        __syncthreads();

        const float bias = dw_b[c0 + c_local];
        for (int wi = 0; wi < 4; wi++) {
            int w = wsub + 8 * wi;
            if (w < 28) {
                float accv = 0.f, msum = 0.f, fs = 0.f;
                #pragma unroll
                for (int ky = 0; ky < 7; ky++) {
                    #pragma unroll
                    for (int kx = 0; kx < 7; kx++) {
                        float wv = wsh[c_local][ky * 7 + kx];
                        float aw = fabsf(wv);
                        fs += aw;                       // full_sum over ALL taps
                        int xi = w + kx - 3;
                        if (xi >= 0 && xi < 28) {
                            float a = act[ky][xi];      // broadcast within wave
                            msum += a * aw;
                            accv += a * xs[c_local][ky][xi] * wv;
                        }
                    }
                }
                float a0 = act[3][w];
                float o = (accv / (msum + 1e-5f) * fs + bias) * a0;
                cw[((size_t)b * 784 + hh * 28 + w) * CDIM + c0 + c_local] = o;
            }
        }
        __syncthreads();
    }
}

// ---------------------------------------------------------------------------
// Sparse LayerNorm over C=384 per pixel.  One wave (32 lanes) per pixel.
// ---------------------------------------------------------------------------
__global__ __launch_bounds__(256) void k_ln(
    const float* __restrict__ cw, const int* __restrict__ mask,
    const float* __restrict__ ln_w, const float* __restrict__ ln_b,
    bf16_t* __restrict__ h0)
{
    const int wid = threadIdx.x >> 5, lane = threadIdx.x & 31;
    const int m = blockIdx.x * 8 + wid;
    const float* row = cw + (size_t)m * CDIM;

    float v[12], s = 0.f, s2 = 0.f;
    #pragma unroll
    for (int j = 0; j < 12; j++) {
        v[j] = row[lane + 32 * j];
        s += v[j]; s2 += v[j] * v[j];
    }
    #pragma unroll
    for (int off = 16; off > 0; off >>= 1) {
        s  += __shfl_xor(s,  off);
        s2 += __shfl_xor(s2, off);
    }
    float mu  = s * (1.f / CDIM);
    float var = s2 * (1.f / CDIM) - mu * mu;
    float rs  = rsqrtf(var + 1e-6f);

    int b = m / 784, p = m - b * 784;
    int hh = p / 28, ww = p - hh * 28;
    float a0 = (float)mask[b * 49 + (hh >> 2) * 7 + (ww >> 2)];

    #pragma unroll
    for (int j = 0; j < 12; j++) {
        int c = lane + 32 * j;
        float o = ((v[j] - mu) * rs * ln_w[c] + ln_b[c]) * a0;
        h0[(size_t)m * CDIM + c] = (bf16_t)o;
    }
}

// ---------------------------------------------------------------------------
// Weight swizzle: f32 (K,N) row-major -> bf16 fragment order:
// out[((k/32)*N + n)*32 + k%32]
// ---------------------------------------------------------------------------
__global__ __launch_bounds__(256) void k_swizzle(
    const float* __restrict__ w, bf16_t* __restrict__ out, int K, int N)
{
    int i = blockIdx.x * 256 + threadIdx.x;
    if (i >= K * N) return;
    int k = i / N, n = i - k * N;
    out[(((size_t)(k >> 5) * N + n) << 5) + (k & 31)] = (bf16_t)w[i];
}

// ---------------------------------------------------------------------------
// GEMM1:  (25088 x 384) x (384 x 1536) + bias, exact GELU, bf16 out.
// Block 256 thr (8 waves, 2M x 4N), block tile 64 x 256, K step 32.
// Double-buffered LDS, async global->LDS copies, 1 barrier per K step.
// ---------------------------------------------------------------------------
__global__ __launch_bounds__(256) void k_gemm1(
    const bf16_t* __restrict__ A, const bf16_t* __restrict__ Bw,
    const float* __restrict__ bias, bf16_t* __restrict__ H)
{
    __shared__ bf16_t As[2][64 * LDT];
    __shared__ bf16_t Bs[2][256 * LDT];

    const int t = threadIdx.x, wid = t >> 5, lane = t & 31;
    const int waveM = wid >> 2, waveN = wid & 3;
    const int m0 = blockIdx.x * 64, n0 = blockIdx.y * 256;
    const int hi = lane >> 4, lm = lane & 15;

    const int rowA = t >> 2, kqA = t & 3;

    auto issue = [&](int k0, int buf) {
        copy16(&A[(size_t)(m0 + rowA) * CDIM + k0 + kqA * 8],
               &As[buf][rowA * LDT + kqA * 8]);
        const bf16_t* src = Bw + ((size_t)(k0 >> 5) * HIDD + n0) * 32;
        #pragma unroll
        for (int j = 0; j < 4; j++) {
            int idx = j * 256 + t;              // 16B units
            int n = idx >> 2, kq = idx & 3;
            copy16(&src[idx * 8], &Bs[buf][n * LDT + kq * 8]);
        }
    };

    v8f acc[2][4] = {};
    issue(0, 0);

    for (int s = 0; s < 12; s++) {
        int buf = s & 1;
        wait_async0();                // this wave's tile-s LDS writes complete
        __syncthreads();              // publish tile s; everyone done with s-1
        if (s + 1 < 12) issue((s + 1) * 32, buf ^ 1);

        v16bf af[2], bfm[4];
        #pragma unroll
        for (int i = 0; i < 2; i++) {
            int row = waveM * 32 + i * 16 + lm;
            union { v16bf v; v8bf h[2]; } u;
            u.h[0] = *(const v8bf*)&As[buf][row * LDT + hi * 8];       // k = hi*8 + e
            u.h[1] = *(const v8bf*)&As[buf][row * LDT + 16 + hi * 8];  // k = 16 + hi*8 + e
            af[i] = u.v;
        }
        #pragma unroll
        for (int j = 0; j < 4; j++) {
            int n = waveN * 64 + j * 16 + lm;
            union { v16bf v; v8bf h[2]; } u;
            u.h[0] = *(const v8bf*)&Bs[buf][n * LDT + hi * 8];         // k = hi*8 + e
            u.h[1] = *(const v8bf*)&Bs[buf][n * LDT + 16 + hi * 8];
            bfm[j] = u.v;
        }
        #pragma unroll
        for (int i = 0; i < 2; i++)
            #pragma unroll
            for (int j = 0; j < 4; j++)
                acc[i][j] = __builtin_amdgcn_wmma_f32_16x16x32_bf16(
                    false, af[i], false, bfm[j], (short)0, acc[i][j], false, false);
    }

    #pragma unroll
    for (int i = 0; i < 2; i++)
        #pragma unroll
        for (int j = 0; j < 4; j++) {
            int col = n0 + waveN * 64 + j * 16 + lm;
            float bb = bias[col];
            #pragma unroll
            for (int r = 0; r < 8; r++) {
                int m = m0 + waveM * 32 + i * 16 + r + 8 * hi;
                float v = acc[i][j][r] + bb;
                float g = 0.5f * v * (1.0f + erff(v * 0.70710678118654752f));
                H[(size_t)m * HIDD + col] = (bf16_t)g;
            }
        }
}

// ---------------------------------------------------------------------------
// GRN part 1: sumsq[b][d] = sum_p h[b,p,d]^2   (coalesced, no atomics)
// ---------------------------------------------------------------------------
__global__ __launch_bounds__(256) void k_sumsq(
    const bf16_t* __restrict__ H, float* __restrict__ ss)
{
    int b = blockIdx.x;
    int d = blockIdx.y * 256 + threadIdx.x;
    const bf16_t* base = H + (size_t)b * 784 * HIDD + d;
    float s = 0.f;
    for (int p = 0; p < 784; p++) {
        float v = (float)base[(size_t)p * HIDD];
        s += v * v;
    }
    ss[b * HIDD + d] = s;
}

// ---------------------------------------------------------------------------
// GRN part 2: per-batch scale  sc[b][d] = 1 + grn_g[d]*Gx/(mean(Gx)+eps)
// ---------------------------------------------------------------------------
__global__ __launch_bounds__(256) void k_nx(
    const float* __restrict__ ss, const float* __restrict__ grn_g,
    float* __restrict__ sc)
{
    __shared__ float red[8];
    int b = blockIdx.x, t = threadIdx.x;
    float gx[6], s = 0.f;
    #pragma unroll
    for (int j = 0; j < 6; j++) {
        gx[j] = sqrtf(ss[b * HIDD + t + 256 * j]);
        s += gx[j];
    }
    #pragma unroll
    for (int off = 16; off > 0; off >>= 1) s += __shfl_xor(s, off);
    if ((t & 31) == 0) red[t >> 5] = s;
    __syncthreads();
    float tot = 0.f;
    #pragma unroll
    for (int i = 0; i < 8; i++) tot += red[i];
    float inv = 1.0f / (tot * (1.0f / HIDD) + 1e-6f);
    #pragma unroll
    for (int j = 0; j < 6; j++) {
        int d = t + 256 * j;
        sc[b * HIDD + d] = 1.0f + grn_g[d] * gx[j] * inv;
    }
}

// ---------------------------------------------------------------------------
// GRN part 3: in-place  h = h*sc[b,d] + grn_b[d]   (16B vectorized)
// ---------------------------------------------------------------------------
__global__ __launch_bounds__(256) void k_grn_apply(
    bf16_t* __restrict__ H, const float* __restrict__ sc,
    const float* __restrict__ grn_b)
{
    size_t i = ((size_t)blockIdx.x * 256 + threadIdx.x) * 8;
    size_t m = i / HIDD;
    int d = (int)(i - m * HIDD);
    int b = (int)(m / 784);
    v8bf hv = *(v8bf*)&H[i];
    #pragma unroll
    for (int j = 0; j < 8; j++) {
        float v = (float)hv[j];
        hv[j] = (bf16_t)(v * sc[b * HIDD + d + j] + grn_b[d + j]);
    }
    *(v8bf*)&H[i] = hv;
}

// ---------------------------------------------------------------------------
// GEMM2: (25088 x 1536) x (1536 x 384); epilogue fuses bias, gamma,
// active mask and residual; writes NCHW f32 output.
// Block 256 thr (8 waves, 4M x 2N), block tile 128 x 128, K step 32.
// Double-buffered LDS, async global->LDS copies, 1 barrier per K step.
// ---------------------------------------------------------------------------
__global__ __launch_bounds__(256) void k_gemm2(
    const bf16_t* __restrict__ A, const bf16_t* __restrict__ Bw,
    const float* __restrict__ pw2_b, const float* __restrict__ gamma,
    const int* __restrict__ mask, const float* __restrict__ x,
    float* __restrict__ out)
{
    __shared__ bf16_t As[2][128 * LDT];
    __shared__ bf16_t Bs[2][128 * LDT];

    const int t = threadIdx.x, wid = t >> 5, lane = t & 31;
    const int waveM = wid >> 1, waveN = wid & 1;
    const int m0 = blockIdx.x * 128, n0 = blockIdx.y * 128;
    const int hi = lane >> 4, lm = lane & 15;

    auto issue = [&](int k0, int buf) {
        #pragma unroll
        for (int j = 0; j < 2; j++) {           // A: 128 rows x 32 k
            int idx = j * 256 + t;
            int row = idx >> 2, kq = idx & 3;
            copy16(&A[(size_t)(m0 + row) * HIDD + k0 + kq * 8],
                   &As[buf][row * LDT + kq * 8]);
        }
        const bf16_t* src = Bw + ((size_t)(k0 >> 5) * CDIM + n0) * 32;
        #pragma unroll
        for (int j = 0; j < 2; j++) {           // B: contiguous 8KB
            int idx = j * 256 + t;
            int n = idx >> 2, kq = idx & 3;
            copy16(&src[idx * 8], &Bs[buf][n * LDT + kq * 8]);
        }
    };

    v8f acc[2][4] = {};
    issue(0, 0);

    for (int s = 0; s < 48; s++) {
        int buf = s & 1;
        wait_async0();
        __syncthreads();
        if (s + 1 < 48) issue((s + 1) * 32, buf ^ 1);

        v16bf af[2], bfm[4];
        #pragma unroll
        for (int i = 0; i < 2; i++) {
            int row = waveM * 32 + i * 16 + lm;
            union { v16bf v; v8bf h[2]; } u;
            u.h[0] = *(const v8bf*)&As[buf][row * LDT + hi * 8];
            u.h[1] = *(const v8bf*)&As[buf][row * LDT + 16 + hi * 8];
            af[i] = u.v;
        }
        #pragma unroll
        for (int j = 0; j < 4; j++) {
            int n = waveN * 64 + j * 16 + lm;
            union { v16bf v; v8bf h[2]; } u;
            u.h[0] = *(const v8bf*)&Bs[buf][n * LDT + hi * 8];
            u.h[1] = *(const v8bf*)&Bs[buf][n * LDT + 16 + hi * 8];
            bfm[j] = u.v;
        }
        #pragma unroll
        for (int i = 0; i < 2; i++)
            #pragma unroll
            for (int j = 0; j < 4; j++)
                acc[i][j] = __builtin_amdgcn_wmma_f32_16x16x32_bf16(
                    false, af[i], false, bfm[j], (short)0, acc[i][j], false, false);
    }

    #pragma unroll
    for (int i = 0; i < 2; i++)
        #pragma unroll
        for (int j = 0; j < 4; j++) {
            int c = n0 + waveN * 64 + j * 16 + lm;
            float bb = pw2_b[c], gm = gamma[c];
            #pragma unroll
            for (int r = 0; r < 8; r++) {
                int m = m0 + waveM * 32 + i * 16 + r + 8 * hi;
                int b = m / 784, p = m - b * 784;
                int hh = p / 28, ww = p - hh * 28;
                float a0 = (float)mask[b * 49 + (hh >> 2) * 7 + (ww >> 2)];
                size_t xi = ((size_t)b * CDIM + c) * 784 + p;
                out[xi] = x[xi] + a0 * gm * (acc[i][j][r] + bb);
            }
        }
}

// ---------------------------------------------------------------------------
extern "C" void kernel_launch(void* const* d_in, const int* in_sizes, int n_in,
                              void* d_out, int out_size, void* d_ws, size_t ws_size,
                              hipStream_t stream) {
    const float* x     = (const float*)d_in[0];
    const int*   mask  = (const int*)  d_in[1];
    const float* dw_w  = (const float*)d_in[2];
    const float* dw_b  = (const float*)d_in[3];
    const float* ln_w  = (const float*)d_in[4];
    const float* ln_b  = (const float*)d_in[5];
    const float* pw1_w = (const float*)d_in[6];
    const float* pw1_b = (const float*)d_in[7];
    const float* grn_g = (const float*)d_in[8];
    const float* grn_b = (const float*)d_in[9];
    const float* pw2_w = (const float*)d_in[10];
    const float* pw2_b = (const float*)d_in[11];
    const float* gamma = (const float*)d_in[12];
    float* out = (float*)d_out;

    char* ws = (char*)d_ws;
    float*  conv_out = (float*) (ws + 0);            // 38,535,168 B
    bf16_t* h0       = (bf16_t*)(ws + 38535168);     // 19,267,584 B
    bf16_t* H        = (bf16_t*)(ws + 57802752);     // 77,070,336 B
    bf16_t* w1s      = (bf16_t*)(ws + 134873088);    //  1,179,648 B
    bf16_t* w2s      = (bf16_t*)(ws + 136052736);    //  1,179,648 B
    float*  ss       = (float*) (ws + 137232384);    //    196,608 B
    float*  sc       = (float*) (ws + 137428992);    //    196,608 B

    k_swizzle<<<2304, 256, 0, stream>>>(pw1_w, w1s, CDIM, HIDD);
    k_swizzle<<<2304, 256, 0, stream>>>(pw2_w, w2s, HIDD, CDIM);
    k_dwconv<<<dim3(32, 28), 256, 0, stream>>>(x, mask, dw_w, dw_b, conv_out);
    k_ln<<<3136, 256, 0, stream>>>(conv_out, mask, ln_w, ln_b, h0);
    k_gemm1<<<dim3(392, 6), 256, 0, stream>>>(h0, w1s, pw1_b, H);
    k_sumsq<<<dim3(32, 6), 256, 0, stream>>>(H, ss);
    k_nx<<<32, 256, 0, stream>>>(ss, grn_g, sc);
    k_grn_apply<<<18816, 256, 0, stream>>>(H, sc, grn_b);
    k_gemm2<<<dim3(196, 3), 256, 0, stream>>>(H, w2s, pw2_b, gamma, mask, x, out);
}
